// DeltaNet_19430432047178
// MI455X (gfx1250) — matrix operand order repaired and checked by reading, not compiled, not run on hardware
//
#include <hip/hip_runtime.h>
#include <cstdint>
#include <cstddef>

// ---------------------------------------------------------------------------
// DeltaNet-ish block for MI455X (gfx1250, wave32, WMMA).
// All big GEMMs run through v_wmma_f32_16x16x32_bf16 (fp32 accumulate).
// Weight streams are L2-resident (192MB L2); inner loops use direct global
// fragment loads + global_prefetch to hide L2 latency.
// ---------------------------------------------------------------------------

#define T_SEQ   2048
#define DMODEL  1024
#define NHEADS  16
#define HDIM    64
#define NEXP    4
#define HIDDEN_ 4096
#define CHUNK   64
#define NCHUNK  (T_SEQ / CHUNK)

typedef __bf16  v16bf __attribute__((ext_vector_type(16)));
typedef float   v8f   __attribute__((ext_vector_type(8)));

union FragAB { v16bf v; uint32_t u[8]; };
union FragC  { v8f   v; float    f[8]; };

__device__ __forceinline__ uint16_t f2bf(float f) {
    uint32_t u = __float_as_uint(f);
    uint32_t r = (u + 0x7FFFu + ((u >> 16) & 1u)) >> 16;   // RNE
    return (uint16_t)r;
}

// ---------------------------------------------------------------------------
// fp32 -> bf16 elementwise convert (weights & activations)
// ---------------------------------------------------------------------------
__global__ void cvt_bf16_kernel(const float* __restrict__ in,
                                uint16_t* __restrict__ out, int n) {
    int i = blockIdx.x * blockDim.x + threadIdx.x;
    int stride = gridDim.x * blockDim.x;
    for (; i < n; i += stride) out[i] = f2bf(in[i]);
}

// ---------------------------------------------------------------------------
// Depthwise causal conv3 (writes bf16 A-matrix for the pointwise GEMM)
// ---------------------------------------------------------------------------
__global__ void conv_kernel(const float* __restrict__ x,
                            const float* __restrict__ w_dw,
                            uint16_t* __restrict__ convb) {
    int id = blockIdx.x * blockDim.x + threadIdx.x;
    if (id >= T_SEQ * DMODEL) return;
    int t = id / DMODEL, d = id % DMODEL;
    float w0 = w_dw[d * 3 + 0], w1 = w_dw[d * 3 + 1], w2 = w_dw[d * 3 + 2];
    float a = (t >= 2) ? x[(t - 2) * DMODEL + d] : 0.f;
    float b = (t >= 1) ? x[(t - 1) * DMODEL + d] : 0.f;
    float c = x[t * DMODEL + d];
    convb[id] = f2bf(a * w0 + b * w1 + c * w2);
}

// ---------------------------------------------------------------------------
// Tiled WMMA GEMM:  C[M,N] = A[M,K] @ W[N,K]^T   (bf16 in, fp32 accumulate)
// Block: 128 threads = 4 waves; wave computes 32 rows x 64 cols (2x4 frags).
// Epilogues (template EPI):
//   0: outF = aux0 + acc ; outB = bf16(outF)            (pw + residual)
//   1: outF = acc                                        (plain)
//   2: outF = aux0 + sigmoid(acc + bias[col]) * aux1     (attn gate -> x2)
//   3: outB[e] = bf16(silu(acc + bias[e,col]))           (MoE w1, grid.z = e)
// ---------------------------------------------------------------------------
template <int EPI>
__global__ void __launch_bounds__(128)
gemm_bf16_kernel(const uint16_t* __restrict__ A,
                 const uint16_t* __restrict__ W,
                 int M, int N, int K,
                 float* __restrict__ outF,
                 uint16_t* __restrict__ outB,
                 const float* __restrict__ aux0,
                 const float* __restrict__ aux1,
                 const float* __restrict__ bias) {
    const int wave  = threadIdx.x >> 5;
    const int lane  = threadIdx.x & 31;
    const int lhalf = lane >> 4;     // 0 or 1
    const int lmod  = lane & 15;

    if constexpr (EPI == 3) {        // per-expert bases
        size_t e = blockIdx.z;
        W    += e * (size_t)N * (size_t)K;
        bias += e * (size_t)N;
        outB += e * (size_t)M * (size_t)N;
    }

    const int rowBase = blockIdx.x * 128 + wave * 32;
    const int colBase = blockIdx.y * 64;

    FragC acc[2][4];
#pragma unroll
    for (int ri = 0; ri < 2; ++ri)
#pragma unroll
        for (int ci = 0; ci < 4; ++ci)
#pragma unroll
            for (int r = 0; r < 8; ++r) acc[ri][ci].f[r] = 0.f;

    const uint32_t* A32 = (const uint32_t*)A;
    const uint32_t* W32 = (const uint32_t*)W;

    for (int k0 = 0; k0 < K; k0 += 32) {
        // Prefetch next cacheline of the streamed operands (global_prefetch_b8)
        if (k0 + 64 < K) {
#pragma unroll
            for (int ci = 0; ci < 4; ++ci) {
                int col = colBase + ci * 16 + lmod;
                __builtin_prefetch(&W[(size_t)col * K + k0 + 64], 0, 1);
            }
            __builtin_prefetch(&A[(size_t)(rowBase + lmod) * K + k0 + 64], 0, 1);
            __builtin_prefetch(&A[(size_t)(rowBase + 16 + lmod) * K + k0 + 64], 0, 1);
        }
        FragAB a[2];
#pragma unroll
        for (int ri = 0; ri < 2; ++ri) {
            int row = rowBase + ri * 16 + lmod;
            size_t base = ((size_t)row * K + k0) >> 1;   // dword index
#pragma unroll
            for (int d = 0; d < 8; ++d) {
                // A k order used here (contiguous, not the ISA table order): vgpr d -> K=lhalf*16+2d (matches B)
                int kh = lhalf * 8 + d;
                a[ri].u[d] = A32[base + kh];
            }
        }
        FragAB b[4];
#pragma unroll
        for (int ci = 0; ci < 4; ++ci) {
            int col = colBase + ci * 16 + lmod;
            // B k order used here (contiguous, same as A): lanes 0-15 K=0..15, lanes 16-31 K=16..31
            size_t base = ((size_t)col * K + k0 + lhalf * 16) >> 1;
#pragma unroll
            for (int d = 0; d < 8; ++d) b[ci].u[d] = W32[base + d];
        }
#pragma unroll
        for (int ri = 0; ri < 2; ++ri)
#pragma unroll
            for (int ci = 0; ci < 4; ++ci)
                acc[ri][ci].v = __builtin_amdgcn_wmma_f32_16x16x32_bf16(
                    false, a[ri].v, false, b[ci].v, (short)0, acc[ri][ci].v,
                    false, false);
    }

    // epilogue: element (r, lane) -> row = base + r + 8*lhalf, col = base + lmod
#pragma unroll
    for (int ri = 0; ri < 2; ++ri) {
#pragma unroll
        for (int r = 0; r < 8; ++r) {
            int row = rowBase + ri * 16 + r + lhalf * 8;
#pragma unroll
            for (int ci = 0; ci < 4; ++ci) {
                int col = colBase + ci * 16 + lmod;
                size_t idx = (size_t)row * N + col;
                float v = acc[ri][ci].f[r];
                if constexpr (EPI == 0) {
                    float o = aux0[idx] + v;
                    outF[idx] = o;
                    outB[idx] = f2bf(o);
                } else if constexpr (EPI == 1) {
                    outF[idx] = v;
                } else if constexpr (EPI == 2) {
                    float g = 1.f / (1.f + __expf(-(v + bias[col])));
                    outF[idx] = aux0[idx] + g * aux1[idx];
                } else if constexpr (EPI == 3) {
                    float s = v + bias[col];
                    outB[idx] = f2bf(s / (1.f + __expf(-s)));   // silu
                }
            }
        }
    }
}

// ---------------------------------------------------------------------------
// Fused MoE down-projection + expert-gated combine + residual:
//   out[t,d] = x2[t,d] + sum_e gw[t,e] * ( hid[e,t,:] @ w2[e,d,:] + b2[e,d] )
// Deterministic (no atomics): each tile loops over all 4 experts.
// ---------------------------------------------------------------------------
__global__ void __launch_bounds__(128)
gemm_moe2_kernel(const uint16_t* __restrict__ hid,   // [E][T][HIDDEN] bf16
                 const uint16_t* __restrict__ W2,    // [E][DMODEL][HIDDEN] bf16
                 const float* __restrict__ gw,       // [T][E]
                 const float* __restrict__ b2,       // [E][DMODEL]
                 const float* __restrict__ x2,       // [T][DMODEL]
                 float* __restrict__ out) {
    constexpr int M = T_SEQ, N = DMODEL, K = HIDDEN_;
    const int wave  = threadIdx.x >> 5;
    const int lane  = threadIdx.x & 31;
    const int lhalf = lane >> 4;
    const int lmod  = lane & 15;
    const int rowBase = blockIdx.x * 128 + wave * 32;
    const int colBase = blockIdx.y * 64;

    FragC fin[2][4];
#pragma unroll
    for (int ri = 0; ri < 2; ++ri)
#pragma unroll
        for (int ci = 0; ci < 4; ++ci)
#pragma unroll
            for (int r = 0; r < 8; ++r) fin[ri][ci].f[r] = 0.f;

    for (int e = 0; e < NEXP; ++e) {
        const uint16_t* Ae = hid + (size_t)e * M * K;
        const uint16_t* We = W2 + (size_t)e * N * K;
        const uint32_t* A32 = (const uint32_t*)Ae;
        const uint32_t* W32 = (const uint32_t*)We;
        FragC acc[2][4];
#pragma unroll
        for (int ri = 0; ri < 2; ++ri)
#pragma unroll
            for (int ci = 0; ci < 4; ++ci)
#pragma unroll
                for (int r = 0; r < 8; ++r) acc[ri][ci].f[r] = 0.f;

        for (int k0 = 0; k0 < K; k0 += 32) {
            if (k0 + 64 < K) {
#pragma unroll
                for (int ci = 0; ci < 4; ++ci) {
                    int col = colBase + ci * 16 + lmod;
                    __builtin_prefetch(&We[(size_t)col * K + k0 + 64], 0, 1);
                }
                __builtin_prefetch(&Ae[(size_t)(rowBase + lmod) * K + k0 + 64], 0, 1);
                __builtin_prefetch(&Ae[(size_t)(rowBase + 16 + lmod) * K + k0 + 64], 0, 1);
            }
            FragAB a[2];
#pragma unroll
            for (int ri = 0; ri < 2; ++ri) {
                int row = rowBase + ri * 16 + lmod;
                size_t base = ((size_t)row * K + k0) >> 1;
#pragma unroll
                for (int d = 0; d < 8; ++d) {
                    int kh = lhalf * 8 + d;
                    a[ri].u[d] = A32[base + kh];
                }
            }
            FragAB b[4];
#pragma unroll
            for (int ci = 0; ci < 4; ++ci) {
                int col = colBase + ci * 16 + lmod;
                size_t base = ((size_t)col * K + k0 + lhalf * 16) >> 1;
#pragma unroll
                for (int d = 0; d < 8; ++d) b[ci].u[d] = W32[base + d];
            }
#pragma unroll
            for (int ri = 0; ri < 2; ++ri)
#pragma unroll
                for (int ci = 0; ci < 4; ++ci)
                    acc[ri][ci].v = __builtin_amdgcn_wmma_f32_16x16x32_bf16(
                        false, a[ri].v, false, b[ci].v, (short)0,
                        acc[ri][ci].v, false, false);
        }
        // weight by gate and accumulate
#pragma unroll
        for (int ri = 0; ri < 2; ++ri) {
#pragma unroll
            for (int r = 0; r < 8; ++r) {
                int row = rowBase + ri * 16 + r + lhalf * 8;
                float g = gw[row * NEXP + e];
#pragma unroll
                for (int ci = 0; ci < 4; ++ci) {
                    int col = colBase + ci * 16 + lmod;
                    fin[ri][ci].f[r] += g * (acc[ri][ci].f[r] + b2[e * N + col]);
                }
            }
        }
    }
#pragma unroll
    for (int ri = 0; ri < 2; ++ri)
#pragma unroll
        for (int r = 0; r < 8; ++r) {
            int row = rowBase + ri * 16 + r + lhalf * 8;
#pragma unroll
            for (int ci = 0; ci < 4; ++ci) {
                int col = colBase + ci * 16 + lmod;
                size_t idx = (size_t)row * N + col;
                out[idx] = x2[idx] + fin[ri][ci].f[r];
            }
        }
}

// ---------------------------------------------------------------------------
// RoPE + phi = elu+1 on q,k (in-place on the qkv fp32 buffer [T][3][H][Dh])
// ---------------------------------------------------------------------------
__global__ void qkv_post_kernel(float* __restrict__ qkv) {
    int id = blockIdx.x * blockDim.x + threadIdx.x;       // T*H*32 pairs
    if (id >= T_SEQ * NHEADS * (HDIM / 2)) return;
    int p = id % (HDIM / 2);
    int h = (id / (HDIM / 2)) % NHEADS;
    int t = id / (NHEADS * (HDIM / 2));
    float inv_freq = __powf(10000.f, -(float)(2 * p) / (float)HDIM);
    float th = (float)t * inv_freq;
    float s = __sinf(th), c = __cosf(th);
#pragma unroll
    for (int comp = 0; comp < 2; ++comp) {                // 0=q, 1=k
        size_t base = (size_t)t * 3 * DMODEL + comp * DMODEL + h * HDIM + 2 * p;
        float x1 = qkv[base], x2 = qkv[base + 1];
        float r1 = x1 * c - x2 * s;
        float r2 = x1 * s + x2 * c;
        // phi = elu(x)+1
        r1 = (r1 > 0.f) ? (r1 + 1.f) : __expf(r1);
        r2 = (r2 > 0.f) ? (r2 + 1.f) : __expf(r2);
        qkv[base] = r1; qkv[base + 1] = r2;
    }
}

// ---------------------------------------------------------------------------
// Linear attention, chunked scan. Pass A: per-chunk sums of k (x) v and k.
// ---------------------------------------------------------------------------
__global__ void __launch_bounds__(256)
attn_chunk_kernel(const float* __restrict__ qkv,
                  float* __restrict__ chunkKV,  // [NCHUNK*H][64][64]
                  float* __restrict__ chunkK) { // [NCHUNK*H][64]
    __shared__ float lk[CHUNK][HDIM];
    __shared__ float lv[CHUNK][HDIM];
    int c = blockIdx.x / NHEADS, h = blockIdx.x % NHEADS;
    int tid = threadIdx.x;
    for (int idx = tid; idx < CHUNK * HDIM; idx += 256) {
        int tok = idx >> 6, j = idx & 63;
        size_t base = (size_t)(c * CHUNK + tok) * 3 * DMODEL + h * HDIM + j;
        lk[tok][j] = qkv[base + DMODEL];
        lv[tok][j] = qkv[base + 2 * DMODEL];
    }
    __syncthreads();
    int j = tid & 63, i0 = (tid >> 6) * 16;
    float s[16];
#pragma unroll
    for (int ii = 0; ii < 16; ++ii) s[ii] = 0.f;
    for (int tok = 0; tok < CHUNK; ++tok) {
        float vj = lv[tok][j];
#pragma unroll
        for (int ii = 0; ii < 16; ++ii) s[ii] += lk[tok][i0 + ii] * vj;
    }
    size_t cb = (size_t)blockIdx.x * HDIM * HDIM;
#pragma unroll
    for (int ii = 0; ii < 16; ++ii) chunkKV[cb + (i0 + ii) * HDIM + j] = s[ii];
    if (tid < HDIM) {
        float ks = 0.f;
        for (int tok = 0; tok < CHUNK; ++tok) ks += lk[tok][tid];
        chunkK[(size_t)blockIdx.x * HDIM + tid] = ks;
    }
}

// Pass B: exclusive scan across chunks (per head), in place.
__global__ void attn_scan_kernel(float* __restrict__ chunkKV,
                                 float* __restrict__ chunkK) {
    int id = blockIdx.x * blockDim.x + threadIdx.x;
    if (id < NHEADS * HDIM * HDIM) {
        int h = id / (HDIM * HDIM), off = id % (HDIM * HDIM);
        float acc = 0.f;
        for (int c = 0; c < NCHUNK; ++c) {
            size_t idx = (size_t)(c * NHEADS + h) * (HDIM * HDIM) + off;
            float t = chunkKV[idx];
            chunkKV[idx] = acc;
            acc += t;
        }
    } else if (id < NHEADS * HDIM * HDIM + NHEADS * HDIM) {
        int id2 = id - NHEADS * HDIM * HDIM;
        int h = id2 / HDIM, off = id2 % HDIM;
        float acc = 0.f;
        for (int c = 0; c < NCHUNK; ++c) {
            size_t idx = (size_t)(c * NHEADS + h) * HDIM + off;
            float t = chunkK[idx];
            chunkK[idx] = acc;
            acc += t;
        }
    }
}

// Pass C: in-chunk inclusive scan. Thread j owns state column S[:,j] (64 regs).
__global__ void __launch_bounds__(64)
attn_inner_kernel(const float* __restrict__ qkv,
                  const float* __restrict__ chunkKV,
                  const float* __restrict__ chunkK,
                  float* __restrict__ ao_num,   // [T][DMODEL] (h*64+j)
                  float* __restrict__ ksumF) {  // [T][DMODEL]
    __shared__ float lq[CHUNK][HDIM];
    __shared__ float lk[CHUNK][HDIM];
    __shared__ float lv[CHUNK][HDIM];
    int c = blockIdx.x / NHEADS, h = blockIdx.x % NHEADS;
    int j = threadIdx.x;
    for (int idx = j; idx < CHUNK * HDIM; idx += 64) {
        int tok = idx >> 6, jj = idx & 63;
        size_t base = (size_t)(c * CHUNK + tok) * 3 * DMODEL + h * HDIM + jj;
        lq[tok][jj] = qkv[base];
        lk[tok][jj] = qkv[base + DMODEL];
        lv[tok][jj] = qkv[base + 2 * DMODEL];
    }
    __syncthreads();
    float S[HDIM];
    size_t cb = (size_t)blockIdx.x * HDIM * HDIM;
#pragma unroll
    for (int i = 0; i < HDIM; ++i) S[i] = chunkKV[cb + i * HDIM + j];
    float ksum = chunkK[(size_t)blockIdx.x * HDIM + j];

    for (int tok = 0; tok < CHUNK; ++tok) {
        float vj = lv[tok][j];
        ksum += lk[tok][j];
        float num = 0.f;
#pragma unroll 8
        for (int i = 0; i < HDIM; ++i) {
            S[i] += lk[tok][i] * vj;      // inclusive state update
            num += lq[tok][i] * S[i];
        }
        int t = c * CHUNK + tok;
        size_t o = (size_t)t * DMODEL + h * HDIM + j;
        ao_num[o] = num;
        ksumF[o]  = ksum;
    }
}

// denom = q_phi . k_prefix ; normalize and emit bf16 A-matrix for w_o GEMM.
__global__ void __launch_bounds__(64)
attn_denorm_kernel(const float* __restrict__ qkv,
                   const float* __restrict__ ksumF,
                   const float* __restrict__ ao_num,
                   uint16_t* __restrict__ aob) {
    __shared__ float red[64];
    int t = blockIdx.x / NHEADS, h = blockIdx.x % NHEADS;
    int j = threadIdx.x;
    size_t o = (size_t)t * DMODEL + h * HDIM + j;
    float q = qkv[(size_t)t * 3 * DMODEL + h * HDIM + j];
    red[j] = q * ksumF[o];
    __syncthreads();
    for (int s = 32; s > 0; s >>= 1) {
        if (j < s) red[j] += red[j + s];
        __syncthreads();
    }
    float denom = red[0] + 1e-6f;
    aob[o] = f2bf(ao_num[o] / denom);
}

// delta[t] = x1[t] - x1[t-1] (0 at t==0), bf16
__global__ void delta_kernel(const float* __restrict__ x1,
                             uint16_t* __restrict__ delta_b) {
    int id = blockIdx.x * blockDim.x + threadIdx.x;
    if (id >= T_SEQ * DMODEL) return;
    int t = id / DMODEL, d = id % DMODEL;
    float prev = (t > 0) ? x1[(t - 1) * DMODEL + d] : x1[d];
    delta_b[id] = f2bf(x1[id] - prev);
}

// LayerNorm + MoE softmax gate (fused, one block per token)
__global__ void __launch_bounds__(256)
ln_gate_kernel(const float* __restrict__ x2,
               const float* __restrict__ ln_g, const float* __restrict__ ln_b,
               const float* __restrict__ wmg,
               float* __restrict__ h, uint16_t* __restrict__ hb,
               float* __restrict__ gw) {
    __shared__ float red[256];
    __shared__ float logits[NEXP];
    int t = blockIdx.x, tid = threadIdx.x;
    float xs[4], sum = 0.f;
#pragma unroll
    for (int k = 0; k < 4; ++k) {
        xs[k] = x2[(size_t)t * DMODEL + tid + k * 256];
        sum += xs[k];
    }
    red[tid] = sum; __syncthreads();
    for (int s = 128; s > 0; s >>= 1) { if (tid < s) red[tid] += red[tid + s]; __syncthreads(); }
    float mu = red[0] * (1.f / DMODEL); __syncthreads();
    float vs = 0.f;
#pragma unroll
    for (int k = 0; k < 4; ++k) { float d = xs[k] - mu; vs += d * d; }
    red[tid] = vs; __syncthreads();
    for (int s = 128; s > 0; s >>= 1) { if (tid < s) red[tid] += red[tid + s]; __syncthreads(); }
    float rstd = rsqrtf(red[0] * (1.f / DMODEL) + 1e-5f); __syncthreads();
    float hv[4];
#pragma unroll
    for (int k = 0; k < 4; ++k) {
        int d = tid + k * 256;
        hv[k] = (xs[k] - mu) * rstd * ln_g[d] + ln_b[d];
        h[(size_t)t * DMODEL + d] = hv[k];
        hb[(size_t)t * DMODEL + d] = f2bf(hv[k]);
    }
    for (int e = 0; e < NEXP; ++e) {
        float p = 0.f;
#pragma unroll
        for (int k = 0; k < 4; ++k) p += hv[k] * wmg[e * DMODEL + tid + k * 256];
        red[tid] = p; __syncthreads();
        for (int s = 128; s > 0; s >>= 1) { if (tid < s) red[tid] += red[tid + s]; __syncthreads(); }
        if (tid == 0) logits[e] = red[0];
        __syncthreads();
    }
    if (tid == 0) {
        float m = logits[0];
        for (int e = 1; e < NEXP; ++e) m = fmaxf(m, logits[e]);
        float se = 0.f, ex[NEXP];
        for (int e = 0; e < NEXP; ++e) { ex[e] = __expf(logits[e] - m); se += ex[e]; }
        for (int e = 0; e < NEXP; ++e) gw[t * NEXP + e] = ex[e] / se;
    }
}

// ---------------------------------------------------------------------------
// Host orchestration. Workspace use ~240 MB (bump-allocated from d_ws).
// ---------------------------------------------------------------------------
extern "C" void kernel_launch(void* const* d_in, const int* in_sizes, int n_in,
                              void* d_out, int out_size, void* d_ws, size_t ws_size,
                              hipStream_t stream) {
    const float* x        = (const float*)d_in[0];
    const float* w_dw     = (const float*)d_in[1];
    const float* w_pw     = (const float*)d_in[2];
    const float* w_qkv    = (const float*)d_in[3];
    const float* w_o      = (const float*)d_in[4];
    const float* w_gate   = (const float*)d_in[5];
    const float* b_gate   = (const float*)d_in[6];
    const float* ln_g     = (const float*)d_in[7];
    const float* ln_b     = (const float*)d_in[8];
    const float* w_moe    = (const float*)d_in[9];
    const float* w1       = (const float*)d_in[10];
    const float* b1       = (const float*)d_in[11];
    const float* w2       = (const float*)d_in[12];
    const float* b2       = (const float*)d_in[13];
    float* out = (float*)d_out;

    char* ws = (char*)d_ws;
    size_t off = 0;
    auto alloc = [&](size_t bytes) -> void* {
        void* p = ws + off;
        off += (bytes + 255) & ~(size_t)255;
        return p;
    };

    uint16_t* wb_pw   = (uint16_t*)alloc((size_t)DMODEL * DMODEL * 2);
    uint16_t* wb_qkv  = (uint16_t*)alloc((size_t)3 * DMODEL * DMODEL * 2);
    uint16_t* wb_o    = (uint16_t*)alloc((size_t)DMODEL * DMODEL * 2);
    uint16_t* wb_gate = (uint16_t*)alloc((size_t)DMODEL * DMODEL * 2);
    uint16_t* wb1     = (uint16_t*)alloc((size_t)NEXP * HIDDEN_ * DMODEL * 2);
    uint16_t* wb2     = (uint16_t*)alloc((size_t)NEXP * DMODEL * HIDDEN_ * 2);
    uint16_t* convb   = (uint16_t*)alloc((size_t)T_SEQ * DMODEL * 2);
    float*    x1      = (float*)   alloc((size_t)T_SEQ * DMODEL * 4);
    uint16_t* x1b     = (uint16_t*)alloc((size_t)T_SEQ * DMODEL * 2);
    float*    qkvbuf  = (float*)   alloc((size_t)T_SEQ * 3 * DMODEL * 4);
    float*    chunkKV = (float*)   alloc((size_t)NCHUNK * NHEADS * HDIM * HDIM * 4);
    float*    chunkK  = (float*)   alloc((size_t)NCHUNK * NHEADS * HDIM * 4);
    float*    ao_num  = (float*)   alloc((size_t)T_SEQ * DMODEL * 4);
    float*    ksumF   = (float*)   alloc((size_t)T_SEQ * DMODEL * 4);
    uint16_t* aob     = (uint16_t*)alloc((size_t)T_SEQ * DMODEL * 2);
    float*    aproj   = (float*)   alloc((size_t)T_SEQ * DMODEL * 4);
    uint16_t* delta_b = (uint16_t*)alloc((size_t)T_SEQ * DMODEL * 2);
    float*    x2      = (float*)   alloc((size_t)T_SEQ * DMODEL * 4);
    float*    hbuf    = (float*)   alloc((size_t)T_SEQ * DMODEL * 4);
    uint16_t* hb      = (uint16_t*)alloc((size_t)T_SEQ * DMODEL * 2);
    float*    gw      = (float*)   alloc((size_t)T_SEQ * NEXP * 4);
    uint16_t* hid     = (uint16_t*)alloc((size_t)NEXP * T_SEQ * HIDDEN_ * 2);
    (void)ws_size; (void)n_in; (void)in_sizes; (void)out_size;

    auto cvt = [&](const float* src, uint16_t* dst, size_t n) {
        int blocks = (int)((n + 255) / 256);
        if (blocks > 16384) blocks = 16384;
        cvt_bf16_kernel<<<blocks, 256, 0, stream>>>(src, dst, (int)n);
    };
    cvt(w_pw,   wb_pw,   (size_t)DMODEL * DMODEL);
    cvt(w_qkv,  wb_qkv,  (size_t)3 * DMODEL * DMODEL);
    cvt(w_o,    wb_o,    (size_t)DMODEL * DMODEL);
    cvt(w_gate, wb_gate, (size_t)DMODEL * DMODEL);
    cvt(w1,     wb1,     (size_t)NEXP * HIDDEN_ * DMODEL);
    cvt(w2,     wb2,     (size_t)NEXP * DMODEL * HIDDEN_);

    conv_kernel<<<(T_SEQ * DMODEL + 255) / 256, 256, 0, stream>>>(x, w_dw, convb);

    // x1 = x + conv @ w_pw^T  (also bf16 copy for next GEMM)
    gemm_bf16_kernel<0><<<dim3(T_SEQ / 128, DMODEL / 64, 1), 128, 0, stream>>>(
        convb, wb_pw, T_SEQ, DMODEL, DMODEL, x1, x1b, x, nullptr, nullptr);

    // qkv = x1 @ w_qkv^T
    gemm_bf16_kernel<1><<<dim3(T_SEQ / 128, (3 * DMODEL) / 64, 1), 128, 0, stream>>>(
        x1b, wb_qkv, T_SEQ, 3 * DMODEL, DMODEL, qkvbuf, nullptr, nullptr, nullptr, nullptr);

    qkv_post_kernel<<<(T_SEQ * NHEADS * (HDIM / 2) + 255) / 256, 256, 0, stream>>>(qkvbuf);

    attn_chunk_kernel<<<NCHUNK * NHEADS, 256, 0, stream>>>(qkvbuf, chunkKV, chunkK);
    {
        int tot = NHEADS * HDIM * HDIM + NHEADS * HDIM;
        attn_scan_kernel<<<(tot + 255) / 256, 256, 0, stream>>>(chunkKV, chunkK);
    }
    attn_inner_kernel<<<NCHUNK * NHEADS, 64, 0, stream>>>(qkvbuf, chunkKV, chunkK,
                                                          ao_num, ksumF);
    attn_denorm_kernel<<<T_SEQ * NHEADS, 64, 0, stream>>>(qkvbuf, ksumF, ao_num, aob);

    // attn_proj = ao @ w_o^T
    gemm_bf16_kernel<1><<<dim3(T_SEQ / 128, DMODEL / 64, 1), 128, 0, stream>>>(
        aob, wb_o, T_SEQ, DMODEL, DMODEL, aproj, nullptr, nullptr, nullptr, nullptr);

    delta_kernel<<<(T_SEQ * DMODEL + 255) / 256, 256, 0, stream>>>(x1, delta_b);

    // x2 = x1 + sigmoid(delta @ w_gate^T + b_gate) * attn_proj
    gemm_bf16_kernel<2><<<dim3(T_SEQ / 128, DMODEL / 64, 1), 128, 0, stream>>>(
        delta_b, wb_gate, T_SEQ, DMODEL, DMODEL, x2, nullptr, x1, aproj, b_gate);

    ln_gate_kernel<<<T_SEQ, 256, 0, stream>>>(x2, ln_g, ln_b, w_moe, hbuf, hb, gw);

    // hid[e] = silu(h @ w1[e]^T + b1[e]) (bf16)
    gemm_bf16_kernel<3><<<dim3(T_SEQ / 128, HIDDEN_ / 64, NEXP), 128, 0, stream>>>(
        hb, wb1, T_SEQ, HIDDEN_, DMODEL, nullptr, hid, nullptr, nullptr, b1);

    // out = x2 + sum_e gw[:,e] * (hid[e] @ w2[e]^T + b2[e])
    gemm_moe2_kernel<<<dim3(T_SEQ / 128, DMODEL / 64, 1), 128, 0, stream>>>(
        hid, wb2, gw, b2, x2, out);
}
